// DiGCN_IB_1BN_Sym_46746424050294
// MI455X (gfx1250) — matrix operand order repaired and checked by reading, not compiled, not run on hardware
//
#include <hip/hip_runtime.h>
#include <hip/hip_bf16.h>

// ---------------------------------------------------------------------------
// DiGCN InceptionBlock + sym branch + Conv1d(k=1) + BatchNorm for MI455X.
//
// Roofline: 3 GEMMs (4.9 GFLOP) -> v_wmma_f32_16x16x32_f16, negligible.
// Dominant cost: 5 x 800K-edge scatter-adds of 128-f32 rows. Feature matrices
// (25.6 MB) are L2-resident (192 MB L2), so gathers + global_atomic_add_f32
// run at L2 atomic throughput; HBM traffic is ~100 MB total (~4us @ 23.3TB/s).
// One wave32 per edge, 4 floats/lane. BN stats fused into the final GEMM via
// ds_swizzle lane-pair reduction + per-channel f32 atomics.
// ---------------------------------------------------------------------------

#define GN 50000
#define GE 800000
#define GD 128
#define BN_EPS 1e-5f

typedef __attribute__((ext_vector_type(16))) _Float16 v16h;
typedef __attribute__((ext_vector_type(8)))  float    v8f;

// SWAPX16 lane swizzle: offset[14:10]=xor(0x10), [9:5]=or(0), [4:0]=and(0x1f)
__device__ inline float lane_swap16(float v) {
    int r = __builtin_amdgcn_ds_swizzle(__builtin_bit_cast(int, v), 0x401f);
    return __builtin_bit_cast(float, r);
}

// ---------------------------------------------------------------------------
__global__ void zero_kernel(float* __restrict__ p, int n) {
    int i = blockIdx.x * blockDim.x + threadIdx.x;
    if (i < n) p[i] = 0.0f;
}

// deg = segment_sum(w, row); w==nullptr means all-ones weights
__global__ void degree_kernel(const int* __restrict__ rows,
                              const float* __restrict__ w,
                              float* __restrict__ deg, int nE) {
    int e = blockIdx.x * blockDim.x + threadIdx.x;
    if (e < nE) atomicAdd(&deg[rows[e]], w ? w[e] : 1.0f);
}

// One wave per edge: acc[col] += dis[row]*w*dis[col] * src[row]
__global__ void __launch_bounds__(256)
scatter_kernel(const float* __restrict__ src, const int* __restrict__ ei,
               const float* __restrict__ w, const float* __restrict__ deg,
               float* __restrict__ acc, int nE) {
    const int wid  = (int)((blockIdx.x * blockDim.x + threadIdx.x) >> 5);
    const int lane = threadIdx.x & 31;
    if (wid >= nE) return;
    const int   row = ei[wid];
    const int   col = ei[nE + wid];
    const float ww  = w ? w[wid] : 1.0f;
    const float dr  = deg[row];
    const float dc  = deg[col];
    const float nr  = dr > 0.0f ? rsqrtf(dr) : 0.0f;
    const float nc  = dc > 0.0f ? rsqrtf(dc) : 0.0f;
    const float nrm = nr * ww * nc;

    const float4 v = *(const float4*)(src + (size_t)row * GD + lane * 4);
    float* dst = acc + (size_t)col * GD + lane * 4;
    atomicAdd(dst + 0, nrm * v.x);
    atomicAdd(dst + 1, nrm * v.y);
    atomicAdd(dst + 2, nrm * v.z);
    atomicAdd(dst + 3, nrm * v.w);
}

// ---------------------------------------------------------------------------
// Fused dual GEMM: symx = X @ W1^T ; x0 = X @ W2^T + b2 (also copied to acc).
// Block = 256 thr = 8 waves; wave w computes cols [16w,16w+16) of 16 rows.
__global__ void __launch_bounds__(256)
gemm_dual_kernel(const float* __restrict__ X,  const float* __restrict__ W1,
                 const float* __restrict__ W2, const float* __restrict__ b2,
                 float* __restrict__ out1, float* __restrict__ out2,
                 float* __restrict__ out2b) {
    const int lane = threadIdx.x & 31;
    const int wave = threadIdx.x >> 5;
    const int r0   = blockIdx.x * 16;
    const int c0   = wave * 16;
    const int lm   = lane & 15;
    const bool hi  = (lane & 16) != 0;

    const float* xr = X + (size_t)(r0 + lm) * GD;   // A: row = r0+lm
    const int    col = c0 + lm;                     // B/C: col

    v8f acc1 = {}; v8f acc2 = {};
#pragma unroll
    for (int kb = 0; kb < 4; ++kb) {
        const int base = kb * 32;
        // A fragment (16x32 f16): halves 0-7 <- K kaA.., halves 8-15 <- K kaB..
        const int kaA = base + (hi ? 16 : 0);
        const int kaB = base + (hi ? 24 : 8);
        v16h a;
#pragma unroll
        for (int i = 0; i < 8; ++i) a[i]     = (_Float16)xr[kaA + i];
#pragma unroll
        for (int i = 0; i < 8; ++i) a[8 + i] = (_Float16)xr[kaB + i];

        // B fragment (32x16 f16): lane holds W[col, kbb..kbb+15] (contiguous)
        const int kbb = base + (hi ? 16 : 0);
        const float* w1r = W1 + (size_t)col * GD + kbb;
        v16h b1;
#pragma unroll
        for (int i = 0; i < 16; ++i) b1[i] = (_Float16)w1r[i];
        acc1 = __builtin_amdgcn_wmma_f32_16x16x32_f16(
            false, a, false, b1, (short)0, acc1, false, false);

        const float* w2r = W2 + (size_t)col * GD + kbb;
        v16h b2v;
#pragma unroll
        for (int i = 0; i < 16; ++i) b2v[i] = (_Float16)w2r[i];
        acc2 = __builtin_amdgcn_wmma_f32_16x16x32_f16(
            false, a, false, b2v, (short)0, acc2, false, false);
    }
    const float bias = b2[col];
#pragma unroll
    for (int r = 0; r < 8; ++r) {   // C layout: VGPR r -> M = r + (hi?8:0)
        const size_t idx = (size_t)(r0 + r + (hi ? 8 : 0)) * GD + col;
        out1[idx] = acc1[r];
        const float v = acc2[r] + bias;
        out2[idx]  = v;
        out2b[idx] = v;
    }
}

// Final GEMM: h2 = H @ Wc^T + bc, fused per-channel sum/sum^2 for BatchNorm.
__global__ void __launch_bounds__(256)
gemm_bn_kernel(const float* __restrict__ H, const float* __restrict__ Wc,
               const float* __restrict__ bc, float* __restrict__ out,
               float* __restrict__ sums, float* __restrict__ sums2) {
    const int lane = threadIdx.x & 31;
    const int wave = threadIdx.x >> 5;
    const int r0   = blockIdx.x * 16;
    const int c0   = wave * 16;
    const int lm   = lane & 15;
    const bool hi  = (lane & 16) != 0;

    const float* hr  = H + (size_t)(r0 + lm) * GD;
    const int    col = c0 + lm;

    v8f acc = {};
#pragma unroll
    for (int kb = 0; kb < 4; ++kb) {
        const int base = kb * 32;
        const int kaA = base + (hi ? 16 : 0);
        const int kaB = base + (hi ? 24 : 8);
        v16h a;
#pragma unroll
        for (int i = 0; i < 8; ++i) a[i]     = (_Float16)hr[kaA + i];
#pragma unroll
        for (int i = 0; i < 8; ++i) a[8 + i] = (_Float16)hr[kaB + i];

        const int kbb = base + (hi ? 16 : 0);
        const float* wr = Wc + (size_t)col * GD + kbb;
        v16h b;
#pragma unroll
        for (int i = 0; i < 16; ++i) b[i] = (_Float16)wr[i];
        acc = __builtin_amdgcn_wmma_f32_16x16x32_f16(
            false, a, false, b, (short)0, acc, false, false);
    }
    const float bias = bc[col];
    float s = 0.0f, q = 0.0f;
#pragma unroll
    for (int r = 0; r < 8; ++r) {
        const float v = acc[r] + bias;
        out[(size_t)(r0 + r + (hi ? 8 : 0)) * GD + col] = v;
        s += v; q += v * v;
    }
    // lanes l and l^16 hold the same channel (other 8 rows) -> pair-reduce
    const float sp = lane_swap16(s);
    const float qp = lane_swap16(q);
    if (!hi) {
        atomicAdd(&sums[col],  s + sp);
        atomicAdd(&sums2[col], q + qp);
    }
}

__global__ void bn_stats_kernel(const float* __restrict__ sums,
                                const float* __restrict__ sums2,
                                const float* __restrict__ gamma,
                                const float* __restrict__ beta,
                                float* __restrict__ scale,
                                float* __restrict__ shift) {
    const int c = threadIdx.x;
    const float inv_n = 1.0f / (float)GN;
    const float mean = sums[c] * inv_n;
    const float var  = sums2[c] * inv_n - mean * mean;
    const float sc   = gamma[c] * rsqrtf(var + BN_EPS);
    scale[c] = sc;
    shift[c] = beta[c] - mean * sc;
}

__global__ void bn_apply_kernel(const float* __restrict__ h2,
                                const float* __restrict__ scale,
                                const float* __restrict__ shift,
                                float* __restrict__ out, int n4) {
    int i = blockIdx.x * blockDim.x + threadIdx.x;
    if (i >= n4) return;
    float4 v = ((const float4*)h2)[i];
    const int c = (i * 4) & (GD - 1);
    v.x = v.x * scale[c + 0] + shift[c + 0];
    v.y = v.y * scale[c + 1] + shift[c + 1];
    v.z = v.z * scale[c + 2] + shift[c + 2];
    v.w = v.w * scale[c + 3] + shift[c + 3];
    ((float4*)out)[i] = v;
}

// ---------------------------------------------------------------------------
extern "C" void kernel_launch(void* const* d_in, const int* in_sizes, int n_in,
                              void* d_out, int out_size, void* d_ws, size_t ws_size,
                              hipStream_t stream) {
    (void)in_sizes; (void)n_in; (void)out_size; (void)ws_size;

    const float* x        = (const float*)d_in[0];
    const int*   e_sym    = (const int*)  d_in[1];   // weight = 1
    const int*   e_in     = (const int*)  d_in[2];
    const float* w_in     = (const float*)d_in[3];
    const int*   e_out    = (const int*)  d_in[4];
    const float* w_out    = (const float*)d_in[5];
    const int*   e1       = (const int*)  d_in[6];
    const float* w1       = (const float*)d_in[7];
    const int*   e2       = (const int*)  d_in[8];
    const float* w2       = (const float*)d_in[9];
    const float* W_lin1   = (const float*)d_in[10];
    const float* W_ln     = (const float*)d_in[11];
    const float* b_ln     = (const float*)d_in[12];
    const float* W_conv   = (const float*)d_in[13];
    const float* b_conv   = (const float*)d_in[14];
    const float* gamma    = (const float*)d_in[15];
    const float* beta     = (const float*)d_in[16];
    float*       out      = (float*)d_out;

    // workspace layout (floats): 4 N*128 matrices + 5 degree vecs + stats
    const size_t NF = (size_t)GN * GD;           // 6.4M floats each
    float* symx  = (float*)d_ws;
    float* x0    = symx + NF;
    float* accb  = x0   + NF;                    // x0 + all scatter results
    float* h2    = accb + NF;
    float* degb  = h2   + NF;                    // 5 * GN
    float* sums  = degb + 5 * (size_t)GN;        // 128
    float* sums2 = sums  + GD;                   // 128
    float* scale = sums2 + GD;                   // 128
    float* shift = scale + GD;                   // 128

    // 1) zero degree arrays + BN accumulators
    {
        const int nz = 5 * GN + 2 * GD;
        zero_kernel<<<(nz + 255) / 256, 256, 0, stream>>>(degb, nz);
    }

    // 2) degrees: deg = segment_sum(w, row) for each of the 5 edge lists
    const int degBlocks = (GE + 255) / 256;
    degree_kernel<<<degBlocks, 256, 0, stream>>>(e_sym, nullptr, degb + 0 * GN, GE);
    degree_kernel<<<degBlocks, 256, 0, stream>>>(e_in,  w_in,    degb + 1 * GN, GE);
    degree_kernel<<<degBlocks, 256, 0, stream>>>(e_out, w_out,   degb + 2 * GN, GE);
    degree_kernel<<<degBlocks, 256, 0, stream>>>(e1,    w1,      degb + 3 * GN, GE);
    degree_kernel<<<degBlocks, 256, 0, stream>>>(e2,    w2,      degb + 4 * GN, GE);

    // 3) fused WMMA dual GEMM: symx = x@W_lin1^T ; x0 = x@W_ln^T + b_ln
    //    (x0 also seeds the accumulator: h_pre = x0 + sum of 5 gconvs)
    gemm_dual_kernel<<<GN / 16, 256, 0, stream>>>(x, W_lin1, W_ln, b_ln,
                                                  symx, x0, accb);

    // 4) five scatter-add aggregations into accb (one wave per edge)
    const int scBlocks = (GE * 32 + 255) / 256;  // = 100000
    scatter_kernel<<<scBlocks, 256, 0, stream>>>(symx, e_sym, nullptr, degb + 0 * GN, accb, GE);
    scatter_kernel<<<scBlocks, 256, 0, stream>>>(symx, e_in,  w_in,    degb + 1 * GN, accb, GE);
    scatter_kernel<<<scBlocks, 256, 0, stream>>>(symx, e_out, w_out,   degb + 2 * GN, accb, GE);
    scatter_kernel<<<scBlocks, 256, 0, stream>>>(x0,   e1,    w1,      degb + 3 * GN, accb, GE);
    scatter_kernel<<<scBlocks, 256, 0, stream>>>(x0,   e2,    w2,      degb + 4 * GN, accb, GE);

    // 5) final WMMA GEMM (Conv1d k=1) + fused BN statistics
    gemm_bn_kernel<<<GN / 16, 256, 0, stream>>>(accb, W_conv, b_conv, h2, sums, sums2);

    // 6) BN finalize + apply
    bn_stats_kernel<<<1, GD, 0, stream>>>(sums, sums2, gamma, beta, scale, shift);
    const int n4 = (int)(NF / 4);
    bn_apply_kernel<<<(n4 + 255) / 256, 256, 0, stream>>>(h2, scale, shift, out, n4);
}
